// MetaPathClassifier_22634477650092
// MI455X (gfx1250) — matrix operand order repaired and checked by reading, not compiled, not run on hardware
//
#include <hip/hip_runtime.h>
#include <hip/hip_bf16.h>

// ---------------- types for WMMA ----------------
typedef __attribute__((ext_vector_type(16))) __bf16 v16bf;
typedef __attribute__((ext_vector_type(8)))  float  v8f;

union Frag16 {
    v16bf v;
    uint4 u[2];
    unsigned short s[16];
};

__device__ __forceinline__ unsigned short f2bf(float f) {
    unsigned u = __float_as_uint(f);
    unsigned r = 0x7fffu + ((u >> 16) & 1u);   // round-to-nearest-even
    return (unsigned short)((u + r) >> 16);
}

__device__ __forceinline__ float sigm(float x) { return 1.0f / (1.0f + __expf(-x)); }

// ---------------- problem constants ----------------
#define BB   4096
#define LL   8
#define HH   256
#define CC   8
#define G3   768          // 3*H
#define NT   48           // G3/16 n-tiles
#define KT   8            // H/32  k-tiles
#define HSTR 264          // padded LDS row stride (fp32 h and bf16 h copies)

// =====================================================================
// Kernel 0: pack W (fp32 [768,256]) into bf16 WMMA B-fragment order.
// pack[((nt*KT+kt)*32 + lane)*16 + s] = bf16( W[n,k] )
//   n = nt*16 + (lane&15),  k = kt*32 + (s>=8 ? 16 : 0) + (lane>=16 ? 8 : 0) + (s&7)
// =====================================================================
__global__ void pack_weights_kernel(const float* __restrict__ W,
                                    unsigned short* __restrict__ pack) {
    int tid = blockIdx.x * blockDim.x + threadIdx.x;
    if (tid >= NT * KT * 512) return;
    int frag = tid >> 9;          // nt*KT + kt
    int within = tid & 511;
    int lane = within >> 4;
    int s = within & 15;
    int nt = frag / KT;
    int kt = frag % KT;
    int n = nt * 16 + (lane & 15);
    int k = kt * 32 + ((s >> 3) << 4) + ((lane >> 4) << 3) + (s & 7);
    pack[tid] = f2bf(W[(size_t)n * HH + k]);
}

// =====================================================================
// Kernel 1: heterogeneous gather + per-type projection -> emb bf16 [B*L, H]
// One block (256 threads) per slot.
// =====================================================================
__global__ void embed_project_kernel(const float* __restrict__ paper_x,
                                     const float* __restrict__ author_x,
                                     const float* __restrict__ venue_x,
                                     const float* __restrict__ Wp, const float* __restrict__ bp,
                                     const float* __restrict__ Wa, const float* __restrict__ ba,
                                     const float* __restrict__ Wv, const float* __restrict__ bv,
                                     const int* __restrict__ type_ids,
                                     const int* __restrict__ node_ids,
                                     unsigned short* __restrict__ emb) {
    __shared__ float xs[128];
    int slot = blockIdx.x;
    int t = threadIdx.x;

    int ty  = type_ids[slot];
    int nid = node_ids[slot];

    const float* xrow;
    const float* W;
    const float* bias;
    int D;
    if (ty == 0)      { xrow = paper_x  + (size_t)nid * 128; W = Wp; bias = bp; D = 128; }
    else if (ty == 1) { xrow = author_x + (size_t)nid * 64;  W = Wa; bias = ba; D = 64;  }
    else              { xrow = venue_x  + (size_t)nid * 32;  W = Wv; bias = bv; D = 32;  }

    if (t < D) xs[t] = xrow[t];
    __syncthreads();

    float acc = bias[t];
    const float* wr = W + (size_t)t * D;
    #pragma unroll 4
    for (int d = 0; d < D; ++d) acc += wr[d] * xs[d];

    emb[(size_t)slot * HH + t] = f2bf(acc);
}

// =====================================================================
// Kernel 2: fused GRU (8 steps) + classifier.
// Block = 256 threads (8 waves), handles 16 batch rows.
// Wave w owns gate columns j in [w*32, w*32+32) for all 3 gates:
//   n_tile(gate g, jt) = g*16 + w*2 + jt  -> register-local gate math.
// h kept twice in LDS: fp32 (carry path) + bf16 (WMMA A fragments).
// =====================================================================
__global__ void __launch_bounds__(256)
gru_classify_kernel(const unsigned short* __restrict__ emb,
                    const unsigned short* __restrict__ bih_pack,
                    const unsigned short* __restrict__ bhh_pack,
                    const float* __restrict__ b_ih,
                    const float* __restrict__ b_hh,
                    const float* __restrict__ Wc,
                    const float* __restrict__ bc,
                    const int* __restrict__ lengths,
                    float* __restrict__ out) {
    __shared__ float          hS[16 * HSTR];   // fp32 hidden state
    __shared__ unsigned short hB[16 * HSTR];   // bf16 mirror for WMMA A

    const int tid  = threadIdx.x;
    const int wave = tid >> 5;
    const int lane = tid & 31;
    const int half = lane >> 4;     // 0 or 1
    const int ml   = lane & 15;     // A-matrix row / B-matrix column within tile
    const int hi   = half * 8;      // A-fragment K sub-offset
    const int m0   = blockIdx.x * 16;

    // zero both h copies
    for (int i = tid; i < 16 * HSTR; i += 256) { hS[i] = 0.0f; hB[i] = 0; }

    // per-lane invariants ------------------------------------------------
    int lenv[8];
    #pragma unroll
    for (int r = 0; r < 8; ++r) lenv[r] = lengths[m0 + r + 8 * half];

    float bI[3][2], bH[3][2];
    #pragma unroll
    for (int g = 0; g < 3; ++g)
        #pragma unroll
        for (int jt = 0; jt < 2; ++jt) {
            int n = g * 256 + wave * 32 + jt * 16 + ml;
            bI[g][jt] = b_ih[n];
            bH[g][jt] = b_hh[n];
        }

    const unsigned short* erow = emb + (size_t)(m0 + ml) * (LL * HH);

    __syncthreads();

    // ---------------- recurrence ----------------
    for (int l = 0; l < LL; ++l) {
        // Opaque zero offset: makes packed-weight addresses appear loop-variant
        // so LICM cannot hoist (and then spill) 96 B fragments out of the step loop.
        unsigned lzero = 0;
        asm volatile("" : "+s"(lzero));
        const unsigned short* bih_l = bih_pack + lzero;
        const unsigned short* bhh_l = bhh_pack + lzero;

        v8f accX[3][2];   // emb @ W_ih^T   (this wave's 6 n-tiles)
        v8f accH[3][2];   // h   @ W_hh^T
        #pragma unroll
        for (int g = 0; g < 3; ++g)
            #pragma unroll
            for (int jt = 0; jt < 2; ++jt) { accX[g][jt] = (v8f){}; accH[g][jt] = (v8f){}; }

        #pragma unroll 1
        for (int kt = 0; kt < KT; ++kt) {
            // A fragment from emb (bf16, global): two 16B runs
            Frag16 aE;
            const unsigned short* eb = erow + l * HH + kt * 32;
            aE.u[0] = *(const uint4*)(eb + hi);
            aE.u[1] = *(const uint4*)(eb + 16 + hi);

            // A fragment from bf16 h in LDS: two ds_load_b128, no conversion
            Frag16 aH;
            const unsigned short* hb = hB + ml * HSTR + kt * 32;
            aH.u[0] = *(const uint4*)(hb + hi);
            aH.u[1] = *(const uint4*)(hb + 16 + hi);

            #pragma unroll
            for (int g = 0; g < 3; ++g) {
                #pragma unroll
                for (int jt = 0; jt < 2; ++jt) {
                    const int ntile = g * 16 + wave * 2 + jt;
                    const size_t fb = ((size_t)(ntile * KT + kt) * 32 + lane) * 16;

                    Frag16 bIg;
                    const uint4* p0 = (const uint4*)(bih_l + fb);
                    bIg.u[0] = p0[0]; bIg.u[1] = p0[1];
                    accX[g][jt] = __builtin_amdgcn_wmma_f32_16x16x32_bf16(
                        false, aE.v, false, bIg.v, (short)0, accX[g][jt], false, false);

                    Frag16 bHg;
                    const uint4* p1 = (const uint4*)(bhh_l + fb);
                    bHg.u[0] = p1[0]; bHg.u[1] = p1[1];
                    accH[g][jt] = __builtin_amdgcn_wmma_f32_16x16x32_bf16(
                        false, aH.v, false, bHg.v, (short)0, accH[g][jt], false, false);
                }
            }
        }

        __syncthreads();   // everyone done reading h for this step

        // register-local gate math; D-tile layout: lane holds col n=ml, rows m=r+8*half
        #pragma unroll
        for (int jt = 0; jt < 2; ++jt) {
            const int col = wave * 32 + jt * 16 + ml;
            #pragma unroll
            for (int r = 0; r < 8; ++r) {
                const int m = r + 8 * half;
                float hold = hS[m * HSTR + col];
                float xr = accX[0][jt][r] + bI[0][jt];
                float hr = accH[0][jt][r] + bH[0][jt];
                float xz = accX[1][jt][r] + bI[1][jt];
                float hz = accH[1][jt][r] + bH[1][jt];
                float xn = accX[2][jt][r] + bI[2][jt];
                float hn = accH[2][jt][r] + bH[2][jt];
                float rg = sigm(xr + hr);
                float zg = sigm(xz + hz);
                float ng = tanhf(xn + rg * hn);
                float hnew = (1.0f - zg) * ng + zg * hold;
                float hout = (l < lenv[r]) ? hnew : hold;
                hS[m * HSTR + col] = hout;
                hB[m * HSTR + col] = f2bf(hout);
            }
        }
        __syncthreads();   // h fully updated before next step's GEMM
    }

    // ---------------- classifier: logits = h @ Wc^T + bc ----------------
    if (tid < 16 * CC) {
        int m = tid >> 3;
        int c = tid & 7;
        float acc = bc[c];
        const float* wc = Wc + (size_t)c * HH;
        #pragma unroll 4
        for (int j = 0; j < HH; ++j) acc += hS[m * HSTR + j] * wc[j];
        out[(size_t)(m0 + m) * CC + c] = acc;
    }
}

// =====================================================================
// launcher
// =====================================================================
extern "C" void kernel_launch(void* const* d_in, const int* in_sizes, int n_in,
                              void* d_out, int out_size, void* d_ws, size_t ws_size,
                              hipStream_t stream) {
    const float* paper_x  = (const float*)d_in[0];
    const float* author_x = (const float*)d_in[1];
    const float* venue_x  = (const float*)d_in[2];
    const float* Wp = (const float*)d_in[3];
    const float* bp = (const float*)d_in[4];
    const float* Wa = (const float*)d_in[5];
    const float* ba = (const float*)d_in[6];
    const float* Wv = (const float*)d_in[7];
    const float* bv = (const float*)d_in[8];
    const float* W_ih = (const float*)d_in[9];
    const float* b_ih = (const float*)d_in[10];
    const float* W_hh = (const float*)d_in[11];
    const float* b_hh = (const float*)d_in[12];
    const float* Wc = (const float*)d_in[13];
    const float* bc = (const float*)d_in[14];
    const int* type_ids = (const int*)d_in[15];
    const int* node_ids = (const int*)d_in[16];
    const int* lengths  = (const int*)d_in[17];
    float* out = (float*)d_out;

    // workspace layout
    const size_t embElems  = (size_t)BB * LL * HH;          // bf16
    const size_t packElems = (size_t)NT * KT * 512;         // bf16 per weight
    unsigned short* emb      = (unsigned short*)d_ws;
    unsigned short* bih_pack = emb + embElems;
    unsigned short* bhh_pack = bih_pack + packElems;

    // 0) pack recurrent weights into WMMA B-fragment order (bf16)
    {
        int total = (int)packElems;
        int blocks = (total + 255) / 256;
        pack_weights_kernel<<<blocks, 256, 0, stream>>>(W_ih, bih_pack);
        pack_weights_kernel<<<blocks, 256, 0, stream>>>(W_hh, bhh_pack);
    }

    // 1) gather + typed projection -> emb (bf16)
    embed_project_kernel<<<BB * LL, HH, 0, stream>>>(
        paper_x, author_x, venue_x, Wp, bp, Wa, ba, Wv, bv,
        type_ids, node_ids, emb);

    // 2) fused GRU recurrence + classifier
    gru_classify_kernel<<<BB / 16, 256, 0, stream>>>(
        emb, bih_pack, bhh_pack, b_ih, b_hh, Wc, bc, lengths, out);
}
